// CopulaModel_85237920956698
// MI455X (gfx1250) — matrix operand order repaired and checked by reading, not compiled, not run on hardware
//
#include <hip/hip_runtime.h>
#include <hip/hip_bf16.h>
#include <math.h>
#include <stdint.h>

// ---------------- problem dims ----------------
#define HD     64     // hidden
#define RK     32     // rank
#define TSTEPS 512
#define CDIM   256
#define BDIM   64
#define NSEQ   (BDIM*CDIM)   // 16384
#define NROWS  64            // sequences per workgroup
#define NTH    256           // 8 wave32

// ---------------- LDS strides (bank-conflict padding) ----------------
#define WSTR 72    // f16 halves per weight row (192 rows)
#define ASTR 72    // f16 halves per h-f16 row (64 rows)
#define GSTR 196   // f32 per gate row (192 cols + pad)
#define GHNS 68    // f32 per gh_n row

typedef __attribute__((ext_vector_type(16))) _Float16 v16h;
typedef __attribute__((ext_vector_type(8)))  float    v8f;
typedef __attribute__((ext_vector_type(4)))  unsigned int u32x4;

union FragU { v16h v; u32x4 q[2]; };

__device__ inline v8f wmma32(v16h a, v16h b, v8f c) {
  // D = A(16x32 f16) x B(32x16 f16) + C(16x16 f32)
  return __builtin_amdgcn_wmma_f32_16x16x32_f16(
      false, a, false, b, (short)0, c, false, false);
}

__device__ inline v8f splat8(float v) {
  v8f a;
#pragma unroll
  for (int i = 0; i < 8; ++i) a[i] = v;
  return a;
}

// A fragment: 16x32 f16 tile; lane holds row M=lane%16, K chunks at hi*8 and 16+hi*8
__device__ inline v16h ldA(const _Float16* base, int mtile, int ktile, int lane) {
  const int r  = mtile * 16 + (lane & 15);
  const int hi = lane >> 4;
  const _Float16* p = base + r * ASTR + ktile * 32 + hi * 8;
  FragU f;
  f.q[0] = *(const u32x4*)(p);
  f.q[1] = *(const u32x4*)(p + 16);
  return f.v;
}

// B fragment: 32x16 f16 tile, B[k][j] = W[j][k] (W row-major [192][WSTR] f16)
__device__ inline v16h ldB(const _Float16* w, int ntile, int ktile, int lane) {
  const int j  = ntile * 16 + (lane & 15);
  const int hi = lane >> 4;
  const _Float16* p = w + j * WSTR + ktile * 32 + hi * 8;
  FragU f;
  f.q[0] = *(const u32x4*)(p);
  f.q[1] = *(const u32x4*)(p + 16);
  return f.v;
}

// C/D store: vgpr r -> row mtile*16 + r + 8*hi, col ntile*16 + lane%16
__device__ inline void stAcc(float* dst, int stride, int mtile, int ntile,
                             int lane, v8f acc) {
  const int col = ntile * 16 + (lane & 15);
  const int hi  = lane >> 4;
  float* p = dst + (mtile * 16 + hi * 8) * stride + col;
#pragma unroll
  for (int r = 0; r < 8; ++r) p[r * stride] = acc[r];
}

__device__ inline float sigmf(float x) { return 1.0f / (1.0f + __expf(-x)); }
__device__ inline float tanh_apx(float x) {
  float e = __expf(2.0f * x);
  return 1.0f - 2.0f / (e + 1.0f);   // safe at +/-inf
}
__device__ inline float softplusf(float t) {
  return fmaxf(t, 0.0f) + log1pf(__expf(-fabsf(t)));
}

// -------- LDS budget (dynamic): f16 region then f32 region --------
constexpr int SM_HALF = 3 * 192 * WSTR + 2 * 64 * ASTR;                 // 50688 halves
constexpr int SM_F32  = 2 * 64 * 64 + 64 * GSTR + 64 * GHNS + 64 +
                        3 * 192 + 128 + 64 + 64;                        // 25984 f32
constexpr size_t SMEM_BYTES = (size_t)SM_HALF * 2 + (size_t)SM_F32 * 4; // 205,312 B

// =================== Kernel 1: batched 2-layer GRU ===================
__global__ __launch_bounds__(NTH)
void gru_kernel(const float* __restrict__ x,
                const float* __restrict__ Wih0, const float* __restrict__ Whh0,
                const float* __restrict__ bih0, const float* __restrict__ bhh0,
                const float* __restrict__ Wih1, const float* __restrict__ Whh1,
                const float* __restrict__ bih1, const float* __restrict__ bhh1,
                float* __restrict__ hT) {
  extern __shared__ char smem[];
  _Float16* w0 = (_Float16*)smem;            // Whh0 f16 [192][WSTR]
  _Float16* w1 = w0 + 192 * WSTR;            // Wih1
  _Float16* w2 = w1 + 192 * WSTR;            // Whh1
  _Float16* a0 = w2 + 192 * WSTR;            // f16(h0) [64][ASTR]
  _Float16* a1 = a0 + 64 * ASTR;             // f16(h1)
  float* h0s = (float*)(a1 + 64 * ASTR);     // [64][64]
  float* h1s = h0s + 64 * 64;
  float* g   = h1s + 64 * 64;                // [64][GSTR] gate accumulators
  float* ghn = g + 64 * GSTR;                // [64][GHNS] layer1 gh_n
  float* xb  = ghn + 64 * GHNS;              // [64]
  float* s_bhh0 = xb + 64;                   // [192]
  float* s_wih0 = s_bhh0 + 192;              // [192]
  float* s_bih0 = s_wih0 + 192;              // [192]
  float* s_b1s  = s_bih0 + 192;              // [128] bih1+bhh1 (r,z)
  float* s_b1x  = s_b1s + 128;               // [64]  bih1 n-part
  float* s_b1h  = s_b1x + 64;                // [64]  bhh1 n-part

  const int tid  = threadIdx.x;
  const int lane = tid & 31;
  const int wv   = tid >> 5;        // wave 0..7
  const int row0 = blockIdx.x * NROWS;

  // ---- preload weights (f32 -> f16 padded) and biases; zero states ----
  for (int i = tid; i < 192 * 64; i += NTH) {
    int r = i >> 6, c = i & 63;
    w0[r * WSTR + c] = (_Float16)Whh0[i];
    w1[r * WSTR + c] = (_Float16)Wih1[i];
    w2[r * WSTR + c] = (_Float16)Whh1[i];
  }
  for (int i = tid; i < 192; i += NTH) {
    s_bhh0[i] = bhh0[i];
    s_wih0[i] = Wih0[i];    // W_ih0 is [192][1]
    s_bih0[i] = bih0[i];
  }
  for (int i = tid; i < 128; i += NTH) s_b1s[i] = bih1[i] + bhh1[i];
  for (int i = tid; i < 64; i += NTH) { s_b1x[i] = bih1[128 + i]; s_b1h[i] = bhh1[128 + i]; }
  for (int i = tid; i < 64 * 64; i += NTH) { h0s[i] = 0.0f; h1s[i] = 0.0f; }
  for (int i = tid; i < 64 * ASTR; i += NTH) { a0[i] = (_Float16)0.0f; a1[i] = (_Float16)0.0f; }
  __syncthreads();

  // GEMM tiling: 4 M-tiles x 12 N-tiles; wave -> (mt = wv&3, half = wv>>2)
  const int mt = wv & 3;
  const int nh = wv >> 2;
  const int ln16 = lane & 15;
  // elementwise mapping: column jj fixed per thread, 16 rows
  const int jj = tid & 63;
  const int rg = tid >> 6;
  const float wr = s_wih0[jj],      wz = s_wih0[jj + 64],  wn = s_wih0[jj + 128];
  const float br = s_bih0[jj],      bz = s_bih0[jj + 64],  bn = s_bih0[jj + 128];

  // ---- hoist per-(tile,lane) bias splat values out of the T loop ----
  float bA[6];   // layer0: bhh0
  float bS[6];   // layer1: nt<8 -> combined bih1+bhh1 (r,z); nt>=8 -> bih1 n-part
  float bH[6];   // layer1: nt>=8 -> bhh1 n-part
#pragma unroll
  for (int nn = 0; nn < 6; ++nn) {
    const int nt = nh * 6 + nn;
    bA[nn] = s_bhh0[nt * 16 + ln16];
    bS[nn] = (nt < 8) ? s_b1s[nt * 16 + ln16] : s_b1x[(nt - 8) * 16 + ln16];
    bH[nn] = (nt >= 8) ? s_b1h[(nt - 8) * 16 + ln16] : 0.0f;
  }

  // async x staging setup (waves 0,1 full-EXEC)
  const float* xrow = x + (size_t)(row0 + tid) * TSTEPS;
  const unsigned xb_off = (unsigned)(uintptr_t)(xb + tid);

#pragma unroll 1
  for (int t = 0; t < TSTEPS; ++t) {
    // -------- phase A: async-stage x_t into LDS (ASYNCcnt path) --------
    if (tid < 64) {
      const float* gp = xrow + t;
      asm volatile("global_load_async_to_lds_b32 %0, %1, off"
                   :: "v"(xb_off), "v"(gp) : "memory");
    }

    // -------- phase B/C: layer0 gh0 = h0 @ Whh0^T + bhh0 --------
    // 3 independent accumulator chains per group to fill WMMA hazard slots
    {
      v16h af0 = ldA(a0, mt, 0, lane);
      v16h af1 = ldA(a0, mt, 1, lane);
#pragma unroll
      for (int grp = 0; grp < 2; ++grp) {
        const int ntb = nh * 6 + grp * 3;
        v8f c0 = splat8(bA[grp * 3 + 0]);
        v8f c1 = splat8(bA[grp * 3 + 1]);
        v8f c2 = splat8(bA[grp * 3 + 2]);
        c0 = wmma32(af0, ldB(w0, ntb + 0, 0, lane), c0);
        c1 = wmma32(af0, ldB(w0, ntb + 1, 0, lane), c1);
        c2 = wmma32(af0, ldB(w0, ntb + 2, 0, lane), c2);
        c0 = wmma32(af1, ldB(w0, ntb + 0, 1, lane), c0);
        c1 = wmma32(af1, ldB(w0, ntb + 1, 1, lane), c1);
        c2 = wmma32(af1, ldB(w0, ntb + 2, 1, lane), c2);
        stAcc(g, GSTR, mt, ntb + 0, lane, c0);
        stAcc(g, GSTR, mt, ntb + 1, lane, c1);
        stAcc(g, GSTR, mt, ntb + 2, lane, c2);
      }
    }
    // drain async x before publishing via barrier (no-op for waves 2..7)
    asm volatile("s_wait_asynccnt 0x0" ::: "memory");
    __syncthreads();

    // -------- phase D: layer0 gates, h0 update, f16 copy --------
    for (int rr = 0; rr < 16; ++rr) {
      const int rw = rg * 16 + rr;
      const float xv = xb[rw];
      const float* grow = g + rw * GSTR;
      const float r = sigmf(grow[jj]       + xv * wr + br);
      const float z = sigmf(grow[jj + 64]  + xv * wz + bz);
      const float n = tanh_apx(xv * wn + bn + r * grow[jj + 128]);
      const float hp = h0s[rw * 64 + jj];
      const float hn = n + z * (hp - n);          // (1-z)*n + z*h
      h0s[rw * 64 + jj] = hn;
      a0[rw * ASTR + jj] = (_Float16)hn;
    }
    __syncthreads();

    // -------- phase E/F: layer1 --------
    // cols 0..127 (r,z): acc = b1s + h0new@Wih1^T + h1@Whh1^T (chained WMMA)
    // cols 128..191 (n): gx and gh kept separate (4 independent chains)
    {
      v16h p0 = ldA(a0, mt, 0, lane), p1 = ldA(a0, mt, 1, lane);  // h0 new
      v16h q0 = ldA(a1, mt, 0, lane), q1 = ldA(a1, mt, 1, lane);  // h1 old
      if (nh == 0) {
        // tiles 0..5: all chained, 2 groups of 3 chains
#pragma unroll
        for (int grp = 0; grp < 2; ++grp) {
          const int ntb = grp * 3;
          v8f c0 = splat8(bS[ntb + 0]);
          v8f c1 = splat8(bS[ntb + 1]);
          v8f c2 = splat8(bS[ntb + 2]);
          c0 = wmma32(p0, ldB(w1, ntb + 0, 0, lane), c0);
          c1 = wmma32(p0, ldB(w1, ntb + 1, 0, lane), c1);
          c2 = wmma32(p0, ldB(w1, ntb + 2, 0, lane), c2);
          c0 = wmma32(p1, ldB(w1, ntb + 0, 1, lane), c0);
          c1 = wmma32(p1, ldB(w1, ntb + 1, 1, lane), c1);
          c2 = wmma32(p1, ldB(w1, ntb + 2, 1, lane), c2);
          c0 = wmma32(q0, ldB(w2, ntb + 0, 0, lane), c0);
          c1 = wmma32(q0, ldB(w2, ntb + 1, 0, lane), c1);
          c2 = wmma32(q0, ldB(w2, ntb + 2, 0, lane), c2);
          c0 = wmma32(q1, ldB(w2, ntb + 0, 1, lane), c0);
          c1 = wmma32(q1, ldB(w2, ntb + 1, 1, lane), c1);
          c2 = wmma32(q1, ldB(w2, ntb + 2, 1, lane), c2);
          stAcc(g, GSTR, mt, ntb + 0, lane, c0);
          stAcc(g, GSTR, mt, ntb + 1, lane, c1);
          stAcc(g, GSTR, mt, ntb + 2, lane, c2);
        }
      } else {
        // tiles 6,7: chained pair
        {
          v8f c0 = splat8(bS[0]);
          v8f c1 = splat8(bS[1]);
          c0 = wmma32(p0, ldB(w1, 6, 0, lane), c0);
          c1 = wmma32(p0, ldB(w1, 7, 0, lane), c1);
          c0 = wmma32(p1, ldB(w1, 6, 1, lane), c0);
          c1 = wmma32(p1, ldB(w1, 7, 1, lane), c1);
          c0 = wmma32(q0, ldB(w2, 6, 0, lane), c0);
          c1 = wmma32(q0, ldB(w2, 7, 0, lane), c1);
          c0 = wmma32(q1, ldB(w2, 6, 1, lane), c0);
          c1 = wmma32(q1, ldB(w2, 7, 1, lane), c1);
          stAcc(g, GSTR, mt, 6, lane, c0);
          stAcc(g, GSTR, mt, 7, lane, c1);
        }
        // tiles 8..11: split gx/gh -> 4 independent chains per 2-tile group
#pragma unroll
        for (int grp = 0; grp < 2; ++grp) {
          const int nt0 = 8 + grp * 2, nt1 = nt0 + 1;
          const int nn0 = 2 + grp * 2, nn1 = nn0 + 1;
          v8f x0 = splat8(bS[nn0]);
          v8f x1 = splat8(bS[nn1]);
          v8f h0c = splat8(bH[nn0]);
          v8f h1c = splat8(bH[nn1]);
          x0  = wmma32(p0, ldB(w1, nt0, 0, lane), x0);
          x1  = wmma32(p0, ldB(w1, nt1, 0, lane), x1);
          h0c = wmma32(q0, ldB(w2, nt0, 0, lane), h0c);
          h1c = wmma32(q0, ldB(w2, nt1, 0, lane), h1c);
          x0  = wmma32(p1, ldB(w1, nt0, 1, lane), x0);
          x1  = wmma32(p1, ldB(w1, nt1, 1, lane), x1);
          h0c = wmma32(q1, ldB(w2, nt0, 1, lane), h0c);
          h1c = wmma32(q1, ldB(w2, nt1, 1, lane), h1c);
          stAcc(g, GSTR, mt, nt0, lane, x0);        // gx_n -> g cols 128..191
          stAcc(g, GSTR, mt, nt1, lane, x1);
          stAcc(ghn, GHNS, mt, nt0 - 8, lane, h0c); // gh_n
          stAcc(ghn, GHNS, mt, nt1 - 8, lane, h1c);
        }
      }
    }
    __syncthreads();

    // -------- phase G: layer1 gates, h1 update, f16 copy --------
    for (int rr = 0; rr < 16; ++rr) {
      const int rw = rg * 16 + rr;
      const float* grow = g + rw * GSTR;
      const float r = sigmf(grow[jj]);
      const float z = sigmf(grow[jj + 64]);
      const float n = tanh_apx(grow[jj + 128] + r * ghn[rw * GHNS + jj]);
      const float hp = h1s[rw * 64 + jj];
      const float hn = n + z * (hp - n);
      h1s[rw * 64 + jj] = hn;
      a1[rw * ASTR + jj] = (_Float16)hn;
    }
    __syncthreads();
  }

  // ---- write hT (top-layer final state) ----
  for (int i = tid; i < 64 * 64; i += NTH)
    hT[(size_t)row0 * HD + i] = h1s[i];
}

// =================== Kernel 2: heads + Woodbury likelihood ===================
// sigma = V V^T + diag(d); M = I + V^T D^-1 V (32x32, Cholesky)
// quad  = r^T D^-1 r - || L^-1 (V^T D^-1 r) ||^2
// logdet= sum log d + 2 sum log L_kk
__global__ __launch_bounds__(CDIM)
void head_kernel(const float* __restrict__ hT, const float* __restrict__ y,
                 const float* __restrict__ Wm, const float* __restrict__ bm,
                 const float* __restrict__ Wd, const float* __restrict__ bd,
                 const float* __restrict__ Wv, const float* __restrict__ bv,
                 float* __restrict__ out) {
  __shared__ float s_wv[RK * HD];
  __shared__ float s_wm[HD], s_wd[HD];
  __shared__ float s_v[CDIM][RK + 1];
  __shared__ float s_dinv[CDIM], s_rd[CDIM], s_logd[CDIM], s_rdr[CDIM];
  __shared__ float s_M[RK][RK + 1];
  __shared__ float s_w[RK];

  const int tid = threadIdx.x;   // == channel c
  const int b   = blockIdx.x;

  for (int i = tid; i < RK * HD; i += CDIM) s_wv[i] = Wv[i];
  if (tid < HD) { s_wm[tid] = Wm[tid]; s_wd[tid] = Wd[tid]; }
  __syncthreads();

  {
    const float* h = hT + ((size_t)b * CDIM + tid) * HD;
    float vr[RK];
#pragma unroll
    for (int r = 0; r < RK; ++r) vr[r] = bv[r];
    float mu = bm[0], td = bd[0];
    for (int k = 0; k < HD; ++k) {
      const float hv = h[k];
      mu += s_wm[k] * hv;
      td += s_wd[k] * hv;
#pragma unroll
      for (int r = 0; r < RK; ++r) vr[r] += s_wv[r * HD + k] * hv;
    }
#pragma unroll
    for (int r = 0; r < RK; ++r) s_v[tid][r] = vr[r];
    const float dv = softplusf(td);
    const float di = 1.0f / dv;
    const float resid = y[b * CDIM + tid] - mu;
    s_dinv[tid] = di;
    s_rd[tid]   = resid;
    s_logd[tid] = __logf(dv);
    s_rdr[tid]  = resid * resid * di;
  }
  __syncthreads();

  // M = I + V^T D^-1 V : 1024 entries, 4 per thread
  {
    const int e0 = tid * 4;
    int ii[4], jj2[4];
    float acc[4] = {0.f, 0.f, 0.f, 0.f};
#pragma unroll
    for (int e = 0; e < 4; ++e) { ii[e] = (e0 + e) >> 5; jj2[e] = (e0 + e) & 31; }
    for (int c = 0; c < CDIM; ++c) {
      const float di = s_dinv[c];
#pragma unroll
      for (int e = 0; e < 4; ++e) acc[e] += s_v[c][ii[e]] * s_v[c][jj2[e]] * di;
    }
#pragma unroll
    for (int e = 0; e < 4; ++e)
      s_M[ii[e]][jj2[e]] = acc[e] + ((ii[e] == jj2[e]) ? 1.0f : 0.0f);
  }
  if (tid < RK) {    // w = V^T D^-1 r
    float a = 0.f;
    for (int c = 0; c < CDIM; ++c) a += s_v[c][tid] * s_dinv[c] * s_rd[c];
    s_w[tid] = a;
  }
  __syncthreads();

  // in-place Cholesky (lower) of M
  for (int k = 0; k < RK; ++k) {
    if (tid == 0) s_M[k][k] = sqrtf(s_M[k][k]);
    __syncthreads();
    if (tid > k && tid < RK) s_M[tid][k] /= s_M[k][k];
    __syncthreads();
    if (tid > k && tid < RK) {
      const float lik = s_M[tid][k];
      for (int j2 = k + 1; j2 <= tid; ++j2) s_M[tid][j2] -= lik * s_M[j2][k];
    }
    __syncthreads();
  }

  if (tid == 0) {
    float yv[RK];
    float quad2 = 0.f, ldM = 0.f;
    for (int k = 0; k < RK; ++k) {        // forward solve L y = w
      float s = s_w[k];
      for (int j2 = 0; j2 < k; ++j2) s -= s_M[k][j2] * yv[j2];
      const float yk = s / s_M[k][k];
      yv[k] = yk;
      quad2 += yk * yk;
      ldM += 2.0f * __logf(s_M[k][k]);
    }
    float sumlogd = 0.f, rdr = 0.f;
    for (int c = 0; c < CDIM; ++c) { sumlogd += s_logd[c]; rdr += s_rdr[c]; }
    const float res = (rdr - quad2) + sumlogd + ldM;
    atomicAdd(out, res * (1.0f / (float)BDIM));
  }
}

__global__ void zero_kernel(float* out) { out[0] = 0.0f; }

// =================== launch ===================
extern "C" void kernel_launch(void* const* d_in, const int* in_sizes, int n_in,
                              void* d_out, int out_size, void* d_ws, size_t ws_size,
                              hipStream_t stream) {
  (void)in_sizes; (void)n_in; (void)out_size; (void)ws_size;
  const float* x    = (const float*)d_in[0];
  const float* y    = (const float*)d_in[1];
  const float* Wih0 = (const float*)d_in[2];
  const float* Whh0 = (const float*)d_in[3];
  const float* bih0 = (const float*)d_in[4];
  const float* bhh0 = (const float*)d_in[5];
  const float* Wih1 = (const float*)d_in[6];
  const float* Whh1 = (const float*)d_in[7];
  const float* bih1 = (const float*)d_in[8];
  const float* bhh1 = (const float*)d_in[9];
  const float* Wm   = (const float*)d_in[10];
  const float* bm   = (const float*)d_in[11];
  const float* Wd   = (const float*)d_in[12];
  const float* bd   = (const float*)d_in[13];
  const float* Wv   = (const float*)d_in[14];
  const float* bv   = (const float*)d_in[15];
  float* out = (float*)d_out;
  float* hT  = (float*)d_ws;    // NSEQ * HD floats = 4 MB

  // allow >64KB dynamic LDS (gfx1250 WGP has 320KB); idempotent, capture-safe
  (void)hipFuncSetAttribute((const void*)gru_kernel,
                            hipFuncAttributeMaxDynamicSharedMemorySize,
                            (int)SMEM_BYTES);

  zero_kernel<<<1, 1, 0, stream>>>(out);
  gru_kernel<<<dim3(NSEQ / NROWS), dim3(NTH), SMEM_BYTES, stream>>>(
      x, Wih0, Whh0, bih0, bhh0, Wih1, Whh1, bih1, bhh1, hT);
  head_kernel<<<dim3(BDIM), dim3(CDIM), 0, stream>>>(
      hT, y, Wm, bm, Wd, bd, Wv, bv, out);
}
